// StochasticLSTM_21285857919550
// MI455X (gfx1250) — compile-verified, matrix operand order checked by reading, not run on hardware
//
#include <hip/hip_runtime.h>
#include <hip/hip_bf16.h>
#include <math.h>

// ---- problem constants (match reference) ----
#define T_  256
#define B_  64
#define D_  512
#define H_  512
#define IT_ 10
#define KTOT 1024            // D + H  (concat [x ; h])
#define NTOT 2048            // 4 gates * H
#define KTILES (KTOT / 32)   // 32  (WMMA K = 32 for bf16)
#define NTILES (NTOT / 16)   // 128

typedef __attribute__((ext_vector_type(16))) __bf16 v16bf;
typedef __attribute__((ext_vector_type(8)))  float  v8f;

// ---------------- workspace layout (elements of __bf16) ----------------
#define XBF_ELEMS  ((size_t)T_ * B_ * D_)                       // 8388608  (16 MB)
#define WPK_PER_IT ((size_t)KTILES * NTILES * 512)              // 2097152
#define WPK_ELEMS  ((size_t)IT_ * WPK_PER_IT)                   // 20971520 (40 MB)

// ---------------- zero d_out ----------------
__global__ void k_zero(float* p, int n) {
    int i = blockIdx.x * blockDim.x + threadIdx.x;
    if (i < n) p[i] = 0.0f;
}

// ---------------- f32 -> bf16 input convert ----------------
__global__ void k_cvt_x(const float* __restrict__ x, __bf16* __restrict__ xb, int n) {
    int i = blockIdx.x * blockDim.x + threadIdx.x;
    if (i < n) xb[i] = (__bf16)x[i];
}

// ---------------- pack masked [W|U] into WMMA B-fragment tile layout ----
// packed index = (((it*KTILES + kt)*NTILES + nt)*32 + lane)*16 + j
// lane = khalf*16 + ncol ; element (k = kt*32 + khalf*16 + j , n = nt*16 + ncol)
// column blocks: [0,512)=i-gate, [512,1024)=f, [1024,1536)=o, [1536,2048)=g
__global__ void k_pack_w(const float* __restrict__ zx, const float* __restrict__ zh,
                         const float* __restrict__ Wi, const float* __restrict__ Ui,
                         const float* __restrict__ Wf, const float* __restrict__ Uf,
                         const float* __restrict__ Wo, const float* __restrict__ Uo,
                         const float* __restrict__ Wg, const float* __restrict__ Ug,
                         __bf16* __restrict__ wpk) {
    size_t idx = (size_t)blockIdx.x * blockDim.x + threadIdx.x;
    if (idx >= WPK_ELEMS) return;
    int j    = (int)(idx & 15);
    int lane = (int)((idx >> 4) & 31);
    int nt   = (int)((idx >> 9) & 127);
    int kt   = (int)((idx >> 16) & 31);
    int it   = (int)(idx >> 21);

    int khalf = lane >> 4;
    int ncol  = lane & 15;
    int k = kt * 32 + khalf * 16 + j;     // 0..1023
    int n = nt * 16 + ncol;               // 0..2047
    int gate = n >> 9;                    // 0..3
    int nH   = n & 511;

    float v;
    if (k < D_) {
        const float* W = (gate == 0) ? Wi : (gate == 1) ? Wf : (gate == 2) ? Wo : Wg;
        v = W[(size_t)k * H_ + nH] * zx[(size_t)it * D_ + k];
    } else {
        int kh = k - D_;
        const float* U = (gate == 0) ? Ui : (gate == 1) ? Uf : (gate == 2) ? Uo : Ug;
        v = U[(size_t)kh * H_ + nH] * zh[(size_t)it * H_ + kh];
    }
    wpk[idx] = (__bf16)v;
}

// Branch-free transcendentals on v_exp_f32 (TRANS pipe, co-executes with WMMA).
__device__ __forceinline__ float sigmf(float x)  { return 1.0f / (1.0f + __expf(-x)); }
__device__ __forceinline__ float tanhff(float x) { return 1.0f - 2.0f / (__expf(2.0f * x) + 1.0f); }

// Async DMA: global -> LDS, 16 bytes per lane (ASYNCcnt-tracked).
__device__ __forceinline__ void async_g2l_b128(void* lds_dst, const void* gsrc) {
    unsigned lds_addr = (unsigned)(size_t)lds_dst;            // low 32b of generic = LDS offset
    unsigned long long gaddr = (unsigned long long)(size_t)gsrc;
    asm volatile("global_load_async_to_lds_b128 %0, %1, off"
                 :: "v"(lds_addr), "v"(gaddr) : "memory");
}
__device__ __forceinline__ void wait_asynccnt0() {
    asm volatile("s_wait_asynccnt 0" ::: "memory");
}

// ---------------- recurrent LSTM kernel ----------------
// grid: IT_ * (B_/16) blocks, 1024 threads (32 waves).
// Block = (iteration it, batch rows b0..b0+15), all 256 steps.
// Wave w owns h-columns [w*16, w*16+16): 4 gate tiles (one per gate).
__global__ void __launch_bounds__(1024)
k_lstm(const __bf16* __restrict__ xbf, const __bf16* __restrict__ wpk_all,
       float* __restrict__ out_o, float* __restrict__ out_h, float* __restrict__ out_c) {
    const int it = blockIdx.x >> 2;
    const int b0 = (blockIdx.x & 3) * 16;
    const int wave = threadIdx.x >> 5;
    const int lane = threadIdx.x & 31;
    const int rowA = lane & 15;           // A-fragment row this lane serves
    const int ksel = (lane >> 4) * 16;    // A-fragment K-half offset

    const __bf16* wpk = wpk_all + (size_t)it * WPK_PER_IT;

    __shared__ __bf16 xb[2][16 * D_];     // double-buffered x tile (2 x 16 KB)
    __shared__ __bf16 hb[16 * H_];        // h state tile           (16 KB)

    // zero h: 8192 bf16, 8 per thread
    {
        int base = threadIdx.x * 8;
        #pragma unroll
        for (int q = 0; q < 8; ++q) hb[base + q] = (__bf16)0.0f;
    }

    // kick off async copy of x_0 into buffer 0 (16 B per lane, 1024 lanes = 16 KB)
    {
        const size_t src = (size_t)(0 * B_ + b0) * D_ + threadIdx.x * 8;
        async_g2l_b128(&xb[0][threadIdx.x * 8], xbf + src);
    }
    wait_asynccnt0();
    __syncthreads();

    v8f cfrag = {};                       // cell-state fragment (16x16, f32)
    const int hc0 = wave * 16;            // h-column base owned by this wave
    const float inv_iter = 1.0f / (float)IT_;

    for (int t = 0; t < T_; ++t) {
        const int cur = t & 1;

        // ---- prefetch x_{t+1} into the other buffer; drains at end of step ----
        if (t + 1 < T_) {
            const size_t src = (size_t)((t + 1) * B_ + b0) * D_ + threadIdx.x * 8;
            async_g2l_b128(&xb[cur ^ 1][threadIdx.x * 8], xbf + src);
        }

        // ---- GEMM: [x;h] (16x1024) @ Wcat (1024x2048), this wave's 4 tiles ----
        v8f acc[4];
        #pragma unroll
        for (int g = 0; g < 4; ++g) acc[g] = (v8f){};

        // K phase 1: x part (kt 0..15) — A from x buffer
        for (int kt = 0; kt < 16; ++kt) {
            const v16bf a = *(const v16bf*)(&xb[cur][rowA * D_ + kt * 32 + ksel]);
            #pragma unroll
            for (int g = 0; g < 4; ++g) {
                const int nt = g * 32 + wave;
                const v16bf b = *(const v16bf*)(wpk + ((size_t)(kt * NTILES + nt) * 512 + lane * 16));
                acc[g] = __builtin_amdgcn_wmma_f32_16x16x32_bf16(
                    false, a, false, b, (short)0, acc[g], false, false);
            }
        }
        // K phase 2: h part (kt 16..31) — A from h buffer
        for (int kh = 0; kh < 16; ++kh) {
            const v16bf a = *(const v16bf*)(&hb[rowA * H_ + kh * 32 + ksel]);
            #pragma unroll
            for (int g = 0; g < 4; ++g) {
                const int nt = g * 32 + wave;
                const v16bf b = *(const v16bf*)(wpk + ((size_t)((kh + 16) * NTILES + nt) * 512 + lane * 16));
                acc[g] = __builtin_amdgcn_wmma_f32_16x16x32_bf16(
                    false, a, false, b, (short)0, acc[g], false, false);
            }
        }
        __syncthreads();   // all waves done reading h before it is overwritten

        // ---- nonlinearities, state update, output accumulation ----
        #pragma unroll
        for (int r = 0; r < 8; ++r) {
            const float ig = sigmf(acc[0][r]);
            const float fg = sigmf(acc[1][r]);
            const float og = sigmf(acc[2][r]);
            const float gg = tanhff(acc[3][r]);
            const float cn = fg * cfrag[r] + ig * gg;
            const float hn = og * tanhff(cn);
            cfrag[r] = cn;

            const int row = r + 8 * (lane >> 4);     // C/D layout: M = r + 8*(lane/16)
            const int col = hc0 + (lane & 15);
            hb[row * H_ + col] = (__bf16)hn;         // h for next step

            const size_t gi = (size_t)(t * B_ + b0 + row) * H_ + col;
            atomicAdd(out_o + gi, og * inv_iter);
            atomicAdd(out_h + gi, hn * inv_iter);
            atomicAdd(out_c + gi, cn * inv_iter);
        }
        wait_asynccnt0();  // x_{t+1} landed in LDS
        __syncthreads();   // h + x_{t+1} visible to all waves
    }
}

extern "C" void kernel_launch(void* const* d_in, const int* in_sizes, int n_in,
                              void* d_out, int out_size, void* d_ws, size_t ws_size,
                              hipStream_t stream) {
    const float* input = (const float*)d_in[0];
    const float* zx    = (const float*)d_in[1];
    const float* zh    = (const float*)d_in[2];
    const float* Wi = (const float*)d_in[3];
    const float* Ui = (const float*)d_in[4];
    const float* Wf = (const float*)d_in[5];
    const float* Uf = (const float*)d_in[6];
    const float* Wo = (const float*)d_in[7];
    const float* Uo = (const float*)d_in[8];
    const float* Wg = (const float*)d_in[9];
    const float* Ug = (const float*)d_in[10];

    __bf16* xbf = (__bf16*)d_ws;
    __bf16* wpk = xbf + XBF_ELEMS;

    float* out_o = (float*)d_out;
    float* out_h = out_o + (size_t)T_ * B_ * H_;
    float* out_c = out_h + (size_t)T_ * B_ * H_;

    // 1) zero output accumulators (atomics accumulate mean over ITER)
    k_zero<<<(out_size + 255) / 256, 256, 0, stream>>>((float*)d_out, out_size);
    // 2) convert input to bf16
    k_cvt_x<<<((int)XBF_ELEMS + 255) / 256, 256, 0, stream>>>(input, xbf, (int)XBF_ELEMS);
    // 3) fold dropout masks into weights, pack to WMMA B-tile layout (bf16)
    k_pack_w<<<((int)WPK_ELEMS + 255) / 256, 256, 0, stream>>>(zx, zh, Wi, Ui, Wf, Uf,
                                                               Wo, Uo, Wg, Ug, wpk);
    // 4) recurrence: 10 iters x 4 batch tiles = 40 WGs, 32 waves each
    k_lstm<<<IT_ * (B_ / 16), 1024, 0, stream>>>(xbf, wpk, out_o, out_h, out_c);
}
